// HeteroSAGE_53549652247154
// MI455X (gfx1250) — matrix operand order repaired and checked by reading, not compiled, not run on hardware
//
#include <hip/hip_runtime.h>
#include <hip/hip_bf16.h>
#include <math.h>

typedef __attribute__((ext_vector_type(16))) __bf16 v16bf;
typedef __attribute__((ext_vector_type(8)))  __bf16 v8bf;
typedef __attribute__((ext_vector_type(8)))  float  v8f;

#define HID 128
#define LSTRIDE (HID + 8)          // padded LDS row stride in halves (rows shift 4 banks)
#define NAUTH 100000
#define NPAP  200000
#define NEDGE 1000000
#define NOUT  40

// ---------------------------------------------------------------------------
// B fragment load from LDS (32 bytes, 16B-aligned -> 2x ds_load_b128)
// ---------------------------------------------------------------------------
__device__ __forceinline__ v16bf load_bfrag(const __bf16* p) {
    v8bf b0 = *(const v8bf*)p;
    v8bf b1 = *(const v8bf*)(p + 8);
    v16bf b;
    #pragma unroll
    for (int j = 0; j < 8; ++j) { b[j] = b0[j]; b[8 + j] = b1[j]; }
    return b;
}

// A fragment (16x32 bf16): lane half selects K sub-blocks {0..15} vs {16..31}
template<bool MEAN>
__device__ __forceinline__ v16bf load_afrag(const float* __restrict__ arow, int k0, int half,
                                            float scale) {
    v16bf a;
    int ka = k0 + half * 16;
    #pragma unroll
    for (int j = 0; j < 8; ++j) {
        float va = arow[ka + j];
        if (MEAN) va *= scale;
        a[j] = (__bf16)va;
    }
    #pragma unroll
    for (int j = 0; j < 8; ++j) {
        float va = arow[8 + ka + j];
        if (MEAN) va *= scale;
        a[8 + j] = (__bf16)va;
    }
    return a;
}

// Stage one fp32 [N,K] weight matrix into LDS as bf16 with padded stride
__device__ __forceinline__ void stage_w(__bf16* lw, const float* __restrict__ W, int N, int K,
                                        int lstride) {
    int khalf = K >> 1;
    int npairs = N * khalf;
    for (int p = threadIdx.x; p < npairs; p += 256) {
        int n  = p / khalf;
        int kk = (p - n * khalf) * 2;
        float2 w2 = *(const float2*)(W + (size_t)n * K + kk);
        union { unsigned u; __bf16 h[2]; } pk;
        pk.h[0] = (__bf16)w2.x;
        pk.h[1] = (__bf16)w2.y;
        *(unsigned*)(lw + n * lstride + kk) = pk.u;
    }
}

// ---------------------------------------------------------------------------
// Fill kernel (zero scratch; graph-capture-safe, no memset)
// ---------------------------------------------------------------------------
__global__ void fill0_kernel(float* __restrict__ p, size_t n) {
    size_t i = (size_t)blockIdx.x * blockDim.x + threadIdx.x;
    size_t stride = (size_t)gridDim.x * blockDim.x;
    for (; i < n; i += stride) p[i] = 0.0f;
}

// ---------------------------------------------------------------------------
// Degree count: one thread per edge
// ---------------------------------------------------------------------------
__global__ void deg_kernel(const int* __restrict__ dst, float* __restrict__ deg, int E) {
    int e = blockIdx.x * blockDim.x + threadIdx.x;
    if (e < E) atomicAdd(deg + dst[e], 1.0f);
}

// ---------------------------------------------------------------------------
// Edge scatter: 32 lanes per edge, 4 floats/lane (HID=128), fp32 atomics.
// h rows + agg rows are L2-resident (192MB L2) so this stays on-chip.
// ---------------------------------------------------------------------------
__global__ void scatter_kernel(const int* __restrict__ src, const int* __restrict__ dst,
                               const float* __restrict__ h, float* __restrict__ agg, int E) {
    int t = blockIdx.x * blockDim.x + threadIdx.x;
    int e = t >> 5;
    int lane = t & 31;
    if (e >= E) return;
    int s = src[e], d = dst[e];
    const float4 v = *(const float4*)(h + (size_t)s * HID + lane * 4);
    float* o = agg + (size_t)d * HID + lane * 4;
    atomicAdd(o + 0, v.x);
    atomicAdd(o + 1, v.y);
    atomicAdd(o + 2, v.z);
    atomicAdd(o + 3, v.w);
}

// ---------------------------------------------------------------------------
// Generic WMMA bf16 GEMM (embeddings / heads):
//   C[M,N] = rowscale(A)[M,K] @ W[N,K]^T (+bias) (+relu)
// W staged once per block into LDS as bf16; wave computes 16 x (NT*16) strip;
// B fragments software-pipelined (load-ahead) so ds latency overlaps WMMA.
// ---------------------------------------------------------------------------
template<int NT, bool BIAS, bool RELU, bool MEAN>
__global__ void gemm_bf16_kernel(const float* __restrict__ A, const float* __restrict__ W,
                                 const float* __restrict__ bias, const float* __restrict__ deg,
                                 float* __restrict__ C, int M, int N, int K) {
    extern __shared__ __align__(16) char smem[];
    __bf16* lw = (__bf16*)smem;                 // [N][K+8] bf16
    const int lstride = K + 8;

    stage_w(lw, W, N, K, lstride);
    __syncthreads();

    int lane  = threadIdx.x & 31;
    int wid   = threadIdx.x >> 5;
    int mtile = blockIdx.x * 8 + wid;
    if (mtile * 16 >= M) return;               // after the barrier: safe

    int l16  = lane & 15;
    int half = lane >> 4;

    int m = mtile * 16 + l16;
    const float* arow = A + (size_t)m * K;
    float scale = MEAN ? (1.0f / fmaxf(deg[m], 1.0f)) : 1.0f;

    int boff[NT];                               // clamped LDS row offsets for N tails
    #pragma unroll
    for (int nt = 0; nt < NT; ++nt) {
        int n = nt * 16 + l16;
        if (n >= N) n = N - 1;
        boff[nt] = n * lstride;
    }

    v8f acc[NT];
    #pragma unroll
    for (int nt = 0; nt < NT; ++nt) acc[nt] = (v8f){};

    for (int k0 = 0; k0 < K; k0 += 32) {
        v16bf a = load_afrag<MEAN>(arow, k0, half, scale);
        int kb = k0 + half * 16;
        v16bf bcur = load_bfrag(lw + boff[0] + kb);
        #pragma unroll
        for (int nt = 0; nt < NT; ++nt) {
            v16bf bnext = (nt + 1 < NT) ? load_bfrag(lw + boff[nt + 1] + kb) : bcur;
            acc[nt] = __builtin_amdgcn_wmma_f32_16x16x32_bf16(false, a, false, bcur, (short)0,
                                                              acc[nt], false, false);
            bcur = bnext;
        }
    }

    int mbase = mtile * 16 + half * 8;
    #pragma unroll
    for (int nt = 0; nt < NT; ++nt) {
        int nn = nt * 16 + l16;
        if (nn >= N) continue;
        float bv = BIAS ? bias[nn] : 0.0f;
        #pragma unroll
        for (int r = 0; r < 8; ++r) {
            size_t idx = (size_t)(mbase + r) * N + nn;
            float v = acc[nt][r] + bv;
            if (RELU) v = fmaxf(v, 0.0f);
            C[idx] = v;
        }
    }
}

// ---------------------------------------------------------------------------
// Fused SAGE pair (N = K = HID = 128):
//   C[M,128] (+)= A1@W1^T + rowscale(A2)@W2^T + bias
// A1 = h_dst (self), A2 = agg (neigh, scaled by 1/max(deg,1) -> mean fused).
// Both 128x128 weight matrices staged in LDS (68KB of 320KB WGP LDS).
// Writes C once per pair instead of store+read+store (saves ~200MB/pair).
// ---------------------------------------------------------------------------
template<bool ACC>
__global__ void sage_pair_kernel(const float* __restrict__ A1, const float* __restrict__ W1,
                                 const float* __restrict__ A2, const float* __restrict__ W2,
                                 const float* __restrict__ deg, const float* __restrict__ bias,
                                 float* __restrict__ C, int M) {
    extern __shared__ __align__(16) char smem[];
    __bf16* lw1 = (__bf16*)smem;                       // [128][136]
    __bf16* lw2 = lw1 + HID * LSTRIDE;                 // [128][136]

    stage_w(lw1, W1, HID, HID, LSTRIDE);
    stage_w(lw2, W2, HID, HID, LSTRIDE);
    __syncthreads();

    int lane  = threadIdx.x & 31;
    int wid   = threadIdx.x >> 5;
    int mtile = blockIdx.x * 8 + wid;
    if (mtile * 16 >= M) return;

    int l16  = lane & 15;
    int half = lane >> 4;

    int m = mtile * 16 + l16;
    const float* a1row = A1 + (size_t)m * HID;
    const float* a2row = A2 + (size_t)m * HID;
    float scale = 1.0f / fmaxf(deg[m], 1.0f);

    int boff = l16 * LSTRIDE;                          // subtile nt adds nt*16*LSTRIDE

    v8f acc[8];
    #pragma unroll
    for (int nt = 0; nt < 8; ++nt) acc[nt] = (v8f){};

    for (int k0 = 0; k0 < HID; k0 += 32) {
        int kb = k0 + half * 16;
        // ---- self stream ----
        v16bf a1 = load_afrag<false>(a1row, k0, half, 1.0f);
        v16bf bcur = load_bfrag(lw1 + boff + kb);
        #pragma unroll
        for (int nt = 0; nt < 8; ++nt) {
            v16bf bnext = (nt < 7) ? load_bfrag(lw1 + boff + (nt + 1) * 16 * LSTRIDE + kb) : bcur;
            acc[nt] = __builtin_amdgcn_wmma_f32_16x16x32_bf16(false, a1, false, bcur, (short)0,
                                                              acc[nt], false, false);
            bcur = bnext;
        }
        // ---- neigh (mean) stream ----
        v16bf a2 = load_afrag<true>(a2row, k0, half, scale);
        bcur = load_bfrag(lw2 + boff + kb);
        #pragma unroll
        for (int nt = 0; nt < 8; ++nt) {
            v16bf bnext = (nt < 7) ? load_bfrag(lw2 + boff + (nt + 1) * 16 * LSTRIDE + kb) : bcur;
            acc[nt] = __builtin_amdgcn_wmma_f32_16x16x32_bf16(false, a2, false, bcur, (short)0,
                                                              acc[nt], false, false);
            bcur = bnext;
        }
    }

    int mbase = mtile * 16 + half * 8;
    #pragma unroll
    for (int nt = 0; nt < 8; ++nt) {
        int nn = nt * 16 + l16;
        float bv = bias[nn];
        #pragma unroll
        for (int r = 0; r < 8; ++r) {
            size_t idx = (size_t)(mbase + r) * HID + nn;
            float v = acc[nt][r] + bv;
            if (ACC) v += C[idx];
            C[idx] = v;
        }
    }
}

// ---------------------------------------------------------------------------
// h_new = LayerNorm(relu(out) + prev) * g + b ; one wave per row (128 feats)
// ---------------------------------------------------------------------------
__global__ void relu_resid_ln_kernel(const float* __restrict__ out, float* __restrict__ h,
                                     const float* __restrict__ g, const float* __restrict__ b,
                                     int M) {
    int row = blockIdx.x * 8 + (threadIdx.x >> 5);
    if (row >= M) return;
    int lane = threadIdx.x & 31;
    const float4 vo = *(const float4*)(out + (size_t)row * HID + lane * 4);
    const float4 vp = *(const float4*)(h   + (size_t)row * HID + lane * 4);
    float x0 = fmaxf(vo.x, 0.0f) + vp.x;
    float x1 = fmaxf(vo.y, 0.0f) + vp.y;
    float x2 = fmaxf(vo.z, 0.0f) + vp.z;
    float x3 = fmaxf(vo.w, 0.0f) + vp.w;
    float s = x0 + x1 + x2 + x3;
    #pragma unroll
    for (int off = 16; off; off >>= 1) s += __shfl_xor(s, off, 32);
    float mean = s * (1.0f / HID);
    float d0 = x0 - mean, d1 = x1 - mean, d2 = x2 - mean, d3 = x3 - mean;
    float vs = d0 * d0 + d1 * d1 + d2 * d2 + d3 * d3;
    #pragma unroll
    for (int off = 16; off; off >>= 1) vs += __shfl_xor(vs, off, 32);
    float inv = rsqrtf(vs * (1.0f / HID) + 1e-5f);
    const float4 gg = *(const float4*)(g + lane * 4);
    const float4 bb = *(const float4*)(b + lane * 4);
    float4 r;
    r.x = d0 * inv * gg.x + bb.x;
    r.y = d1 * inv * gg.y + bb.y;
    r.z = d2 * inv * gg.z + bb.z;
    r.w = d3 * inv * gg.w + bb.w;
    *(float4*)(h + (size_t)row * HID + lane * 4) = r;
}

// ---------------------------------------------------------------------------
// In-place log_softmax over 40 columns; one wave per row.
// ---------------------------------------------------------------------------
__global__ void logsoftmax40_kernel(float* __restrict__ x, int M) {
    int row = blockIdx.x * 8 + (threadIdx.x >> 5);
    if (row >= M) return;
    int lane = threadIdx.x & 31;
    float* r = x + (size_t)row * NOUT;
    float a = r[lane];                                  // lanes 0..31 all valid (40 > 32)
    float b = (lane + 32 < NOUT) ? r[lane + 32] : -INFINITY;
    float mx = fmaxf(a, b);
    #pragma unroll
    for (int off = 16; off; off >>= 1) mx = fmaxf(mx, __shfl_xor(mx, off, 32));
    float se = __expf(a - mx) + ((lane + 32 < NOUT) ? __expf(b - mx) : 0.0f);
    #pragma unroll
    for (int off = 16; off; off >>= 1) se += __shfl_xor(se, off, 32);
    float lse = __logf(se) + mx;
    r[lane] = a - lse;
    if (lane + 32 < NOUT) r[lane + 32] = b - lse;
}

// ---------------------------------------------------------------------------
// Host-side helpers
// ---------------------------------------------------------------------------
template<int NT, bool BIAS, bool RELU, bool MEAN>
static void run_gemm(hipStream_t s, const float* A, const float* W, const float* bias,
                     const float* deg, float* C, int M, int N, int K) {
    dim3 grid((M / 16 + 7) / 8);
    size_t shmem = (size_t)N * (K + 8) * sizeof(__bf16);
    gemm_bf16_kernel<NT, BIAS, RELU, MEAN><<<grid, 256, shmem, s>>>(A, W, bias, deg, C, M, N, K);
}

template<bool ACC>
static void run_pair(hipStream_t s, const float* A1, const float* W1, const float* A2,
                     const float* W2, const float* deg, const float* bias, float* C, int M) {
    dim3 grid((M / 16 + 7) / 8);
    size_t shmem = (size_t)2 * HID * LSTRIDE * sizeof(__bf16);   // 69632 B
    sage_pair_kernel<ACC><<<grid, 256, shmem, s>>>(A1, W1, A2, W2, deg, bias, C, M);
}

static void run_fill0(hipStream_t s, float* p, size_t n) {
    int blocks = (int)((n + 255) / 256);
    if (blocks > 4096) blocks = 4096;
    fill0_kernel<<<blocks, 256, 0, s>>>(p, n);
}

extern "C" void kernel_launch(void* const* d_in, const int* in_sizes, int n_in,
                              void* d_out, int out_size, void* d_ws, size_t ws_size,
                              hipStream_t stream) {
    // ---- inputs (top-level insertion order; params flattened JAX-style) ----
    const float* x_author = (const float*)d_in[0];   // [100000,128]
    const float* x_paper  = (const float*)d_in[1];   // [200000,256]
    const float* Wea = (const float*)d_in[2];        // emb_author W [128,128]
    const float* bea = (const float*)d_in[3];
    const float* Wep = (const float*)d_in[4];        // emb_paper W [128,256]
    const float* bep = (const float*)d_in[5];
    // layers: base 6, stride 9; sorted etypes: cites, rev, writes; sorted: W_neigh, W_self, bias
    const float *WnC[2], *WsC[2], *bC[2], *WnR[2], *WsR[2], *bR[2], *WnW[2], *WsW[2], *bW[2];
    for (int l = 0; l < 2; ++l) {
        int base = 6 + 9 * l;
        WnC[l] = (const float*)d_in[base + 0];
        WsC[l] = (const float*)d_in[base + 1];
        bC[l]  = (const float*)d_in[base + 2];
        WnR[l] = (const float*)d_in[base + 3];
        WsR[l] = (const float*)d_in[base + 4];
        bR[l]  = (const float*)d_in[base + 5];
        WnW[l] = (const float*)d_in[base + 6];
        WsW[l] = (const float*)d_in[base + 7];
        bW[l]  = (const float*)d_in[base + 8];
    }
    const float* lnAg = (const float*)d_in[24];
    const float* lnAb = (const float*)d_in[25];
    const float* lnPg = (const float*)d_in[26];
    const float* lnPb = (const float*)d_in[27];
    const float* W1a = (const float*)d_in[28];
    const float* b1a = (const float*)d_in[29];
    const float* W2a = (const float*)d_in[30];       // [40,128]
    const float* b2a = (const float*)d_in[31];
    const float* W1p = (const float*)d_in[32];
    const float* b1p = (const float*)d_in[33];
    const float* W2p = (const float*)d_in[34];
    const float* b2p = (const float*)d_in[35];
    const int* writes_src = (const int*)d_in[36];
    const int* writes_dst = (const int*)d_in[37];
    const int* cites_src  = (const int*)d_in[38];
    const int* cites_dst  = (const int*)d_in[39];
    const int* rev_src    = (const int*)d_in[40];
    const int* rev_dst    = (const int*)d_in[41];

    // ---- workspace carve-up ----
    float* ws = (float*)d_ws;
    size_t o = 0;
    float* hA   = ws + o; o += (size_t)NAUTH * HID;
    float* hP   = ws + o; o += (size_t)NPAP * HID;
    float* outA = ws + o; o += (size_t)NAUTH * HID;
    float* outP = ws + o; o += (size_t)NPAP * HID;
    float* aggA = ws + o; o += (size_t)NAUTH * HID;   // also MLP hidden (author)
    float* aggP = ws + o; o += (size_t)NPAP * HID;    // also MLP hidden (paper)
    float* degPw = ws + o; o += NPAP;
    float* degPc = ws + o; o += NPAP;
    float* degAr = ws + o; o += NAUTH;

    float* d_outA = (float*)d_out;                    // [100000,40]
    float* d_outP = (float*)d_out + (size_t)NAUTH * NOUT;  // [200000,128]

    const int scatterBlocks = (NEDGE * 32) / 256;     // 32 lanes/edge
    const int degBlocks = (NEDGE + 255) / 256;

    // ---- 1. embeddings ----
    run_gemm<8, true, false, false>(stream, x_author, Wea, bea, nullptr, hA, NAUTH, HID, 128);
    run_gemm<8, true, false, false>(stream, x_paper,  Wep, bep, nullptr, hP, NPAP,  HID, 256);

    // ---- 2. degrees (layer-invariant) ----
    run_fill0(stream, degPw, NPAP);
    run_fill0(stream, degPc, NPAP);
    run_fill0(stream, degAr, NAUTH);
    deg_kernel<<<degBlocks, 256, 0, stream>>>(writes_dst, degPw, NEDGE);
    deg_kernel<<<degBlocks, 256, 0, stream>>>(cites_dst,  degPc, NEDGE);
    deg_kernel<<<degBlocks, 256, 0, stream>>>(rev_dst,    degAr, NEDGE);

    // ---- 3. two SAGE layers ----
    for (int l = 0; l < 2; ++l) {
        // paper dst: writes (a->p), fused self+neigh pair (overwrite)
        run_fill0(stream, aggP, (size_t)NPAP * HID);
        scatter_kernel<<<scatterBlocks, 256, 0, stream>>>(writes_src, writes_dst, hA, aggP, NEDGE);
        run_pair<false>(stream, hP, WsW[l], aggP, WnW[l], degPw, bW[l], outP, NPAP);
        // paper dst: cites (p->p), fused pair (accumulate)
        run_fill0(stream, aggP, (size_t)NPAP * HID);
        scatter_kernel<<<scatterBlocks, 256, 0, stream>>>(cites_src, cites_dst, hP, aggP, NEDGE);
        run_pair<true>(stream, hP, WsC[l], aggP, WnC[l], degPc, bC[l], outP, NPAP);
        // author dst: rev (p->a), fused pair (overwrite)
        run_fill0(stream, aggA, (size_t)NAUTH * HID);
        scatter_kernel<<<scatterBlocks, 256, 0, stream>>>(rev_src, rev_dst, hP, aggA, NEDGE);
        run_pair<false>(stream, hA, WsR[l], aggA, WnR[l], degAr, bR[l], outA, NAUTH);
        // residual + relu + LN (updates hA/hP in place after all reads of prev values)
        relu_resid_ln_kernel<<<(NAUTH + 7) / 8, 256, 0, stream>>>(outA, hA, lnAg, lnAb, NAUTH);
        relu_resid_ln_kernel<<<(NPAP + 7) / 8, 256, 0, stream>>>(outP, hP, lnPg, lnPb, NPAP);
    }

    // ---- 4. output heads ----
    run_gemm<8, true, true,  false>(stream, hA,   W1a, b1a, nullptr, aggA, NAUTH, HID, HID);
    run_gemm<3, true, false, false>(stream, aggA, W2a, b2a, nullptr, d_outA, NAUTH, NOUT, HID);
    logsoftmax40_kernel<<<(NAUTH + 7) / 8, 256, 0, stream>>>(d_outA, NAUTH);
    run_gemm<8, true, true,  false>(stream, hP,   W1p, b1p, nullptr, aggP, NPAP, HID, HID);
    run_gemm<8, true, false, false>(stream, aggP, W2p, b2p, nullptr, d_outP, NPAP, HID, HID);
}
